// GAT_13932873909075
// MI455X (gfx1250) — compile-verified, matrix operand order checked by reading, not run on hardware
//
#include <hip/hip_runtime.h>
#include <hip/hip_bf16.h>
#include <math.h>

typedef __attribute__((ext_vector_type(16))) _Float16 v16h;
typedef __attribute__((ext_vector_type(8)))  float    v8f;

#define NN 50000
#define NE 800000
#define F  128
#define NH 4
#define HD 32
#define NG 64
#define SLOPE 0.2f

// ---- WMMA fragment packing (ISA 7.12.2, 16-bit A 16x32 layout) ----
// A packed: [tileM][kblock][lane][16 halves]   B packed: [kblock][tileN][lane][16 halves]
__device__ __forceinline__ int frag_sub(int koff, int& grp) {
  grp = ((koff & 15) >= 8) ? 1 : 0;                  // lane half holds K {g*8..g*8+7, 16+g*8..}
  return (koff < 16) ? (koff - 8 * grp) : (koff - 8 * grp - 8);
}
__device__ __forceinline__ size_t packA_idx(int n, int f) {
  int tm = n >> 4, r = n & 15, kb = f >> 5, grp;
  int idx = frag_sub(f & 31, grp);
  return ((size_t)(tm * 4 + kb) * 32 + (r + 16 * grp)) * 16 + idx;
}
__device__ __forceinline__ size_t packB_idx(int k, int c) {
  int tn = c >> 4, cc = c & 15, kb = k >> 5, grp;
  int idx = frag_sub(k & 31, grp);
  return ((size_t)(kb * 8 + tn) * 32 + (cc + 16 * grp)) * 16 + idx;
}

// ---------- utility ----------
__global__ void k_fill(float* __restrict__ p, float v, int n) {
  int t = blockIdx.x * blockDim.x + threadIdx.x;
  if (t < n) p[t] = v;
}

// h[n,f] = word_embeds[word_ids[n], f]; also write fragment-packed f16 copy
__global__ void k_gather(const int* __restrict__ wid, const float* __restrict__ emb,
                         float* __restrict__ h, _Float16* __restrict__ hp) {
  int t = blockIdx.x * blockDim.x + threadIdx.x;
  if (t >= NN * F) return;
  int n = t >> 7, f = t & 127;
  float v = emb[(size_t)wid[n] * F + f];
  h[t] = v;
  hp[packA_idx(n, f)] = (_Float16)v;
}

// W[k,c] f32 -> fragment-packed f16
__global__ void k_packW(const float* __restrict__ w, _Float16* __restrict__ wp) {
  int t = blockIdx.x * blockDim.x + threadIdx.x;
  if (t >= F * F) return;
  int k = t >> 7, c = t & 127;
  wp[packB_idx(k, c)] = (_Float16)w[t];
}

// ---------- WMMA GEMM: C[M,128] = A[M,128] x B[128,128] ----------
// one wave -> 16x64 strip: 4 accumulators share each A fragment; 16 static v_wmma
__global__ void k_gemm(const v16h* __restrict__ Ap, const v16h* __restrict__ Bp,
                       float* __restrict__ C, int Mtiles) {
  const int lane = threadIdx.x & 31;
  const int wave = blockIdx.x * (blockDim.x >> 5) + (threadIdx.x >> 5);
  const int tm = wave >> 1;
  const int half = wave & 1;                         // columns 0..63 or 64..127
  if (tm >= Mtiles) return;
  const int r = lane & 15, grp = lane >> 4;

  v8f acc[4] = {};
#pragma unroll
  for (int kb = 0; kb < 4; ++kb) {
    v16h a = Ap[(size_t)(tm * 4 + kb) * 32 + lane];  // 2x global_load_b128, coalesced
#pragma unroll
    for (int t = 0; t < 4; ++t) {
      v16h b = Bp[(size_t)(kb * 8 + half * 4 + t) * 32 + lane];
      acc[t] = __builtin_amdgcn_wmma_f32_16x16x32_f16(false, a, false, b,
                                                      (short)0, acc[t], false, false);
    }
  }
#pragma unroll
  for (int t = 0; t < 4; ++t) {
    int cb = (half * 4 + t) * 16 + r;
#pragma unroll
    for (int i = 0; i < 8; ++i)                       // C/D: vgpr i -> M=i + 8*(lane>=16)
      C[(size_t)(tm * 16 + i + 8 * grp) * F + cb] = acc[t][i];
  }
}

// el[n,h] = sum_d feat[n,h,d]*al[h,d]; er likewise
__global__ void k_eler(const float* __restrict__ feat, const float* __restrict__ al,
                       const float* __restrict__ ar, float* __restrict__ el,
                       float* __restrict__ er) {
  int t = blockIdx.x * blockDim.x + threadIdx.x;
  if (t >= NN * NH) return;
  int n = t >> 2, hd = t & 3;
  float a = 0.f, b = 0.f;
#pragma unroll 8
  for (int d = 0; d < HD; ++d) {
    float v = feat[(size_t)n * F + hd * HD + d];
    a += v * al[hd * HD + d];
    b += v * ar[hd * HD + d];
  }
  el[t] = a;
  er[t] = b;
}

__device__ __forceinline__ float lrelu(float v) { return v > 0.f ? v : SLOPE * v; }

// pass 1: segment max of leaky_relu(el[src]+er[dst]) into nmax[dst]
__global__ void k_edge_max(const int* __restrict__ src, const int* __restrict__ dst,
                           const float* __restrict__ el, const float* __restrict__ er,
                           float* __restrict__ nmax) {
  int e = blockIdx.x * blockDim.x + threadIdx.x;
  if (e >= NE) return;
  int s = src[e], d = dst[e];
#pragma unroll
  for (int h = 0; h < NH; ++h) {
    float v = lrelu(el[s * NH + h] + er[d * NH + h]);
    atomicMax(&nmax[d * NH + h], v);
  }
}

// pass 2: ex = exp(e - m[dst]); store per-edge; segment sum into nsum[dst]
__global__ void k_edge_expsum(const int* __restrict__ src, const int* __restrict__ dst,
                              const float* __restrict__ el, const float* __restrict__ er,
                              const float* __restrict__ nmax, float* __restrict__ eex,
                              float* __restrict__ nsum) {
  int e = blockIdx.x * blockDim.x + threadIdx.x;
  if (e >= NE) return;
  int s = src[e], d = dst[e];
#pragma unroll
  for (int h = 0; h < NH; ++h) {
    float v = lrelu(el[s * NH + h] + er[d * NH + h]);
    float m = nmax[d * NH + h];
    if (!isfinite(m)) m = 0.f;
    float x = expf(v - m);
    eex[(size_t)e * NH + h] = x;
    atomicAdd(&nsum[d * NH + h], x);
  }
}

// pass 3: out[dst] += feat[src] * alpha   (32 lanes per edge, 4 feats per lane)
__global__ void k_edge_msg(const int* __restrict__ src, const int* __restrict__ dst,
                           const float* __restrict__ feat, const float* __restrict__ eex,
                           const float* __restrict__ nsum, float* __restrict__ outp) {
  long long t = (long long)blockIdx.x * blockDim.x + threadIdx.x;
  long long e = t >> 5;
  if (e >= NE) return;
  int lane = (int)(t & 31);
  int s = src[e], d = dst[e];
  const float* frow = feat + (size_t)s * F;
  __builtin_prefetch(frow, 0, 1);                     // global_prefetch (L2-resident gather)
  int hd = lane >> 3;                                 // 4 contiguous feats share a head
  float alpha = eex[(size_t)e * NH + hd] / (nsum[d * NH + hd] + 1e-10f);
  int f0 = lane * 4;
#pragma unroll
  for (int i = 0; i < 4; ++i)
    atomicAdd(&outp[(size_t)d * F + f0 + i], frow[f0 + i] * alpha);
}

// elu + write back as next layer's input (f32 row-major + fragment-packed f16)
__global__ void k_elu(const float* __restrict__ acc, float* __restrict__ h,
                      _Float16* __restrict__ hp) {
  int t = blockIdx.x * blockDim.x + threadIdx.x;
  if (t >= NN * F) return;
  float x = acc[t];
  float y = x > 0.f ? x : expm1f(x);
  h[t] = y;
  hp[packA_idx(t >> 7, t & 127)] = (_Float16)y;
}

// per-graph max pool
__global__ void k_pool(const float* __restrict__ h, const int* __restrict__ gid,
                       float* __restrict__ pooled) {
  int t = blockIdx.x * blockDim.x + threadIdx.x;
  if (t >= NN * F) return;
  int n = t >> 7, f = t & 127;
  atomicMax(&pooled[(size_t)gid[n] * F + f], h[t]);
}

// head: logits, sigmoid, BCE mean.  d_out[0]=loss, d_out[1..64]=y_pred
__global__ void k_final(const float* __restrict__ pooled, const float* __restrict__ out_w,
                        const float* __restrict__ out_b, const float* __restrict__ y,
                        float* __restrict__ out) {
  __shared__ float sh[NG];
  int g = threadIdx.x;
  float acc = out_b[0];
#pragma unroll 8
  for (int f = 0; f < F; ++f) {
    float p = pooled[(size_t)g * F + f];
    if (!isfinite(p)) p = 0.f;                        // empty-graph guard
    acc += p * out_w[f];
  }
  float logit = acc;
  float yp = 1.f / (1.f + expf(-logit));
  float term = fmaxf(logit, 0.f) - logit * y[g] + log1pf(expf(-fabsf(logit)));
  sh[g] = term;
  __syncthreads();
  if (g == 0) {
    float s = 0.f;
    for (int i = 0; i < NG; ++i) s += sh[i];
    out[0] = s / (float)NG;
  }
  out[1 + g] = yp;
}

// ---------- host ----------
extern "C" void kernel_launch(void* const* d_in, const int* in_sizes, int n_in,
                              void* d_out, int out_size, void* d_ws, size_t ws_size,
                              hipStream_t stream) {
  const int*   word_ids = (const int*)d_in[0];
  const int*   edge_src = (const int*)d_in[1];
  const int*   edge_dst = (const int*)d_in[2];
  const int*   node_gid = (const int*)d_in[3];
  const float* y_data   = (const float*)d_in[4];
  const float* w_embeds = (const float*)d_in[5];
  const float* Ws[2]    = {(const float*)d_in[6], (const float*)d_in[9]};
  const float* als[2]   = {(const float*)d_in[7], (const float*)d_in[10]};
  const float* ars[2]   = {(const float*)d_in[8], (const float*)d_in[11]};
  const float* out_w    = (const float*)d_in[12];
  const float* out_b    = (const float*)d_in[13];
  float* out = (float*)d_out;

  char* base = (char*)d_ws;
  size_t off = 0;
  auto take = [&](size_t bytes) { void* p = base + off; off = (off + bytes + 255) & ~(size_t)255; return p; };
  float*    h      = (float*)take((size_t)NN * F * 4);
  _Float16* hp     = (_Float16*)take((size_t)NN * F * 2);    // fragment-packed A
  _Float16* wp     = (_Float16*)take((size_t)F * F * 2);     // fragment-packed B
  float*    feat   = (float*)take((size_t)NN * F * 4);
  float*    el     = (float*)take((size_t)NN * NH * 4);
  float*    er     = (float*)take((size_t)NN * NH * 4);
  float*    nmax   = (float*)take((size_t)NN * NH * 4);
  float*    nsum   = (float*)take((size_t)NN * NH * 4);
  float*    eex    = (float*)take((size_t)NE * NH * 4);
  float*    outacc = (float*)take((size_t)NN * F * 4);
  float*    pooled = (float*)take((size_t)NG * F * 4);

  const int TB = 256;
  const float NINF = -__builtin_inff();
  const int Mtiles = NN / 16;                                 // 3125
  const int gemmWaves = Mtiles * 2;                           // 16x64 strip per wave
  const int gemmBlocks = (gemmWaves + (TB / 32) - 1) / (TB / 32);

  k_gather<<<(NN * F + TB - 1) / TB, TB, 0, stream>>>(word_ids, w_embeds, h, hp);

  for (int layer = 0; layer < 2; ++layer) {
    k_packW<<<(F * F + TB - 1) / TB, TB, 0, stream>>>(Ws[layer], wp);
    k_gemm<<<gemmBlocks, TB, 0, stream>>>((const v16h*)hp, (const v16h*)wp, feat, Mtiles);
    k_eler<<<(NN * NH + TB - 1) / TB, TB, 0, stream>>>(feat, als[layer], ars[layer], el, er);

    k_fill<<<(NN * NH + TB - 1) / TB, TB, 0, stream>>>(nmax, NINF, NN * NH);
    k_fill<<<(NN * NH + TB - 1) / TB, TB, 0, stream>>>(nsum, 0.f, NN * NH);
    k_fill<<<(NN * F + TB - 1) / TB, TB, 0, stream>>>(outacc, 0.f, NN * F);

    k_edge_max<<<(NE + TB - 1) / TB, TB, 0, stream>>>(edge_src, edge_dst, el, er, nmax);
    k_edge_expsum<<<(NE + TB - 1) / TB, TB, 0, stream>>>(edge_src, edge_dst, el, er, nmax, eex, nsum);
    k_edge_msg<<<((long long)NE * 32 + TB - 1) / TB, TB, 0, stream>>>(edge_src, edge_dst, feat, eex, nsum, outacc);

    k_elu<<<(NN * F + TB - 1) / TB, TB, 0, stream>>>(outacc, h, hp);
  }

  k_fill<<<(NG * F + TB - 1) / TB, TB, 0, stream>>>(pooled, NINF, NG * F);
  k_pool<<<(NN * F + TB - 1) / TB, TB, 0, stream>>>(h, node_gid, pooled);
  k_final<<<1, NG, 0, stream>>>(pooled, out_w, out_b, y_data, out);
}
